// EncoderBlock_65180423684119
// MI455X (gfx1250) — matrix operand order repaired and checked by reading, not compiled, not run on hardware
//
#include <hip/hip_runtime.h>
#include <hip/hip_bf16.h>
#include <math.h>

typedef unsigned short u16;
typedef unsigned int u32;

typedef __bf16 v16bf __attribute__((ext_vector_type(16)));
typedef float  v8f   __attribute__((ext_vector_type(8)));

#define BB 2
#define SS 2048
#define DD 1024
#define HH 16
#define DKK 64
#define DFFV 4096
#define MM (BB * SS)   // 4096 rows

// ---------- helpers ----------
__device__ __forceinline__ u16 f2bf(float f) {
  u32 u = __builtin_bit_cast(u32, f);
  u32 r = (u + 0x7FFFu + ((u >> 16) & 1u)) >> 16;
  return (u16)r;
}

union FragCast { u32 u[8]; v16bf v; };
__device__ __forceinline__ v16bf as_bf16x16(const u32* s) {
  FragCast f;
#pragma unroll
  for (int i = 0; i < 8; ++i) f.u[i] = s[i];
  return f.v;
}

__device__ __forceinline__ v8f v8f_zero() {
  v8f z = {0.f, 0.f, 0.f, 0.f, 0.f, 0.f, 0.f, 0.f};
  return z;
}

// ---------- fp32 -> bf16 convert (8 elems/thread, vectorized) ----------
__global__ __launch_bounds__(256) void cvt_bf16_kernel(const float* __restrict__ in,
                                                       u16* __restrict__ out, int n8) {
  int i = blockIdx.x * 256 + threadIdx.x;  // index of 8-element group
  if (i < n8) {
    const float4 a = ((const float4*)in)[i * 2 + 0];
    const float4 b = ((const float4*)in)[i * 2 + 1];
    u32 r[4];
    r[0] = (u32)f2bf(a.x) | ((u32)f2bf(a.y) << 16);
    r[1] = (u32)f2bf(a.z) | ((u32)f2bf(a.w) << 16);
    r[2] = (u32)f2bf(b.x) | ((u32)f2bf(b.y) << 16);
    r[3] = (u32)f2bf(b.z) | ((u32)f2bf(b.w) << 16);
    ((uint4*)out)[i] = make_uint4(r[0], r[1], r[2], r[3]);
  }
}

// ---------- layernorm (unbiased std, scalar alpha/beta) -> bf16 ----------
__global__ __launch_bounds__(256) void ln_bf16_kernel(const float* __restrict__ x,
                                                      const float* __restrict__ alpha,
                                                      const float* __restrict__ beta,
                                                      u16* __restrict__ out, int cols) {
  __shared__ float red[16];
  const int row = blockIdx.x;
  const float* xr = x + (size_t)row * cols;
  float s = 0.f, s2 = 0.f;
  for (int c = threadIdx.x; c < cols; c += 256) {
    float v = xr[c];
    s += v;
    s2 += v * v;
  }
#pragma unroll
  for (int m = 16; m >= 1; m >>= 1) {
    s  += __shfl_xor(s, m, 32);
    s2 += __shfl_xor(s2, m, 32);
  }
  const int wid = threadIdx.x >> 5, lane = threadIdx.x & 31;
  if (lane == 0) { red[wid] = s; red[8 + wid] = s2; }
  __syncthreads();
  if (wid == 0) {
    s  = (lane < 8) ? red[lane] : 0.f;
    s2 = (lane < 8) ? red[8 + lane] : 0.f;
#pragma unroll
    for (int m = 4; m >= 1; m >>= 1) {
      s  += __shfl_xor(s, m, 32);
      s2 += __shfl_xor(s2, m, 32);
    }
    if (lane == 0) { red[0] = s; red[1] = s2; }
  }
  __syncthreads();
  s = red[0]; s2 = red[1];
  const float mean = s / (float)cols;
  const float var  = fmaxf((s2 - (float)cols * mean * mean) / (float)(cols - 1), 0.f);
  const float inv  = 1.f / (sqrtf(var) + 1e-6f);
  const float a = alpha[0] * inv;
  const float g = beta[0];
  u16* orow = out + (size_t)row * cols;
  for (int c = threadIdx.x; c < cols; c += 256) orow[c] = f2bf((xr[c] - mean) * a + g);
}

// ---------- BF16 WMMA GEMM: C = A[MxK] * B[KxN] + bias (+resid) ----------
// 256 threads = 8 waves; macro tile 128x128; wave tile 64x32 (4x2 wmma accs).
// Compile-time epilogue flags -> branch-free epilogue, batched residual loads.
#define ASTR 40  // LDS stride (u16) for A tile: 16B-aligned rows, conflict-free frag reads
#define BSTR 34  // LDS stride (u16) for transposed B tile: u32 k-pair reads, 17-bank col stride
template <bool RELU, bool HAS_RES, bool HAS_OUTF, bool HAS_OUTB>
__global__ __launch_bounds__(256) void gemm_bf16_kernel(const u16* __restrict__ A,
                                                        const u16* __restrict__ Bw,
                                                        const float* __restrict__ bias,
                                                        const float* __restrict__ resid,
                                                        float* __restrict__ outf,
                                                        u16* __restrict__ outb,
                                                        int M, int N, int K) {
  __shared__ u16 As[128 * ASTR];
  __shared__ u16 Bt[128 * BSTR];
  const int tid = threadIdx.x;
  const int lane = tid & 31;
  const int wid = tid >> 5;
  const int wm = wid >> 2;  // 0..1
  const int wn = wid & 3;   // 0..3
  const int bm = blockIdx.y * 128;
  const int bn = blockIdx.x * 128;
  const int rl = lane & 15;
  const int kh = lane >> 4;

  // per-thread global tile coordinates for the loads (2 chunks each)
  const int ar0 = tid >> 2, acg0 = tid & 3;                 // A chunk 0
  const int ar1 = (tid + 256) >> 2, acg1 = (tid + 256) & 3; // A chunk 1
  const int bk0 = tid >> 4, bng0 = tid & 15;                // B chunk 0
  const int bk1 = (tid + 256) >> 4, bng1 = (tid + 256) & 15;

  v8f acc[4][2];
#pragma unroll
  for (int m = 0; m < 4; ++m)
#pragma unroll
    for (int n = 0; n < 2; ++n) acc[m][n] = v8f_zero();

  for (int k0 = 0; k0 < K; k0 += 32) {
    // prefetch next-iteration tiles (lowers to global_prefetch_b8)
    const int kn = k0 + 32;
    if (kn < K) {
      __builtin_prefetch(A + (size_t)(bm + ar0) * K + kn + acg0 * 8, 0, 3);
      __builtin_prefetch(A + (size_t)(bm + ar1) * K + kn + acg1 * 8, 0, 3);
      __builtin_prefetch(Bw + (size_t)(kn + bk0) * N + bn + bng0 * 8, 0, 3);
      __builtin_prefetch(Bw + (size_t)(kn + bk1) * N + bn + bng1 * 8, 0, 3);
    }
    __syncthreads();
    // A tile 128x32 -> LDS (row-major, stride ASTR)
    {
      uint4 d0 = *(const uint4*)(A + (size_t)(bm + ar0) * K + k0 + acg0 * 8);
      uint4 d1 = *(const uint4*)(A + (size_t)(bm + ar1) * K + k0 + acg1 * 8);
      *(uint4*)(&As[ar0 * ASTR + acg0 * 8]) = d0;
      *(uint4*)(&As[ar1 * ASTR + acg1 * 8]) = d1;
    }
    // B tile 32x128 -> LDS transposed: Bt[n][k]
    {
      uint4 d0 = *(const uint4*)(Bw + (size_t)(k0 + bk0) * N + bn + bng0 * 8);
      uint4 d1 = *(const uint4*)(Bw + (size_t)(k0 + bk1) * N + bn + bng1 * 8);
      union { uint4 v; u16 s[8]; } u0, u1;
      u0.v = d0; u1.v = d1;
#pragma unroll
      for (int q = 0; q < 8; ++q) Bt[(bng0 * 8 + q) * BSTR + bk0] = u0.s[q];
#pragma unroll
      for (int q = 0; q < 8; ++q) Bt[(bng1 * 8 + q) * BSTR + bk1] = u1.s[q];
    }
    __syncthreads();

    u32 af[4][8], bfrag[2][8];
#pragma unroll
    for (int m = 0; m < 4; ++m) {
      const u16* p = &As[(wm * 64 + m * 16 + rl) * ASTR];
#pragma unroll
      for (int j = 0; j < 4; ++j) {
        af[m][j]     = *(const u32*)(p + kh * 16 + 2 * j);
        af[m][4 + j] = *(const u32*)(p + 8 + kh * 16 + 2 * j);
      }
    }
#pragma unroll
    for (int n = 0; n < 2; ++n) {
      const u16* p = &Bt[(wn * 32 + n * 16 + rl) * BSTR + kh * 16];
#pragma unroll
      for (int j = 0; j < 8; ++j) bfrag[n][j] = *(const u32*)(p + 2 * j);
    }
#pragma unroll
    for (int m = 0; m < 4; ++m)
#pragma unroll
      for (int n = 0; n < 2; ++n)
        acc[m][n] = __builtin_amdgcn_wmma_f32_16x16x32_bf16(
            false, as_bf16x16(af[m]), false, as_bf16x16(bfrag[n]),
            (short)0, acc[m][n], false, false);
  }

  // branch-free epilogue: bias (+relu) (+resid) -> f32 and/or bf16
#pragma unroll
  for (int n = 0; n < 2; ++n) {
    const int gc = bn + wn * 32 + n * 16 + rl;
    const float bv = bias[gc];
#pragma unroll
    for (int m = 0; m < 4; ++m) {
      const int gr0 = bm + wm * 64 + m * 16 + 8 * kh;
      float rv[8];
      if constexpr (HAS_RES) {
#pragma unroll
        for (int j = 0; j < 8; ++j) rv[j] = resid[(size_t)(gr0 + j) * N + gc];
      }
#pragma unroll
      for (int j = 0; j < 8; ++j) {
        const size_t off = (size_t)(gr0 + j) * N + gc;
        float v = acc[m][n][j] + bv;
        if constexpr (RELU) v = fmaxf(v, 0.f);
        if constexpr (HAS_RES) v += rv[j];
        if constexpr (HAS_OUTF) outf[off] = v;
        if constexpr (HAS_OUTB) outb[off] = f2bf(v);
      }
    }
  }
}

// ---------- flash attention: 1 wave / 16-query tile, 32-key chunks ----------
#define QSTR 72  // 16B-aligned rows, 36-bank stride (conflict-free)
#define VSTR 34  // Vt[d][key], u32 key-pair reads
__global__ __launch_bounds__(32) void flash_attn_kernel(const u16* __restrict__ Q,
                                                        const u16* __restrict__ Kb,
                                                        const u16* __restrict__ Vb,
                                                        u16* __restrict__ Ctx) {
  __shared__ u16 Qs[16 * QSTR];
  __shared__ u16 Ks[32 * QSTR];
  __shared__ u16 Vt[64 * VSTR];
  __shared__ u16 Ps[16 * VSTR];

  const int lane = threadIdx.x;
  const int qt = blockIdx.x % (SS / 16);
  const int bh = blockIdx.x / (SS / 16);
  const int h = bh % HH;
  const int b = bh / HH;
  const size_t base = (size_t)b * SS * DD + (size_t)h * DKK;
  const int qbase = qt * 16;
  const int rl = lane & 15;
  const int kh = lane >> 4;

  // load 16x64 Q tile
#pragma unroll
  for (int it = 0; it < 4; ++it) {
    int chunk = lane + it * 32;  // 0..127
    int r = chunk >> 3, cg = chunk & 7;
    uint4 d = *(const uint4*)(Q + base + (size_t)(qbase + r) * DD + cg * 8);
    *(uint4*)(&Qs[r * QSTR + cg * 8]) = d;
  }
  // Q A-frags (d 0..31 and d 32..63); same-wave DS ops are in-order
  u32 qa0[8], qa1[8];
  {
    const u16* p = &Qs[rl * QSTR];
#pragma unroll
    for (int j = 0; j < 4; ++j) {
      qa0[j]     = *(const u32*)(p + kh * 16 + 2 * j);
      qa0[4 + j] = *(const u32*)(p + 8 + kh * 16 + 2 * j);
      qa1[j]     = *(const u32*)(p + 32 + kh * 16 + 2 * j);
      qa1[4 + j] = *(const u32*)(p + 40 + kh * 16 + 2 * j);
    }
  }

  float mrow[8], lrow[8];
  v8f O[4];
#pragma unroll
  for (int j = 0; j < 8; ++j) { mrow[j] = -1e30f; lrow[j] = 0.f; }
#pragma unroll
  for (int t = 0; t < 4; ++t) O[t] = v8f_zero();
  const float scale = 0.125f;  // 1/sqrt(64)

  for (int kc = 0; kc < SS; kc += 32) {
    // prefetch next chunk of K/V
    if (kc + 32 < SS) {
      __builtin_prefetch(Kb + base + (size_t)(kc + 32 + (lane >> 3)) * DD + (lane & 7) * 8, 0, 3);
      __builtin_prefetch(Vb + base + (size_t)(kc + 32 + (lane >> 3)) * DD + (lane & 7) * 8, 0, 3);
    }
    // K chunk 32x64 -> LDS
#pragma unroll
    for (int it = 0; it < 8; ++it) {
      int chunk = lane + it * 32;
      int r = chunk >> 3, cg = chunk & 7;
      uint4 d = *(const uint4*)(Kb + base + (size_t)(kc + r) * DD + cg * 8);
      *(uint4*)(&Ks[r * QSTR + cg * 8]) = d;
    }
    // V chunk transposed -> Vt[d][key]
#pragma unroll
    for (int it = 0; it < 8; ++it) {
      int chunk = lane + it * 32;
      int r = chunk >> 3, cg = chunk & 7;
      uint4 d = *(const uint4*)(Vb + base + (size_t)(kc + r) * DD + cg * 8);
      union { uint4 v; u16 s[8]; } u;
      u.v = d;
#pragma unroll
      for (int q = 0; q < 8; ++q) Vt[(cg * 8 + q) * VSTR + r] = u.s[q];
    }

    // scores: s0 = keys 0..15, s1 = keys 16..31 (K-dim = d, chained 2x k32)
    v8f s0 = v8f_zero(), s1 = v8f_zero();
    u32 bfr[8];
    {
      const u16* p = &Ks[rl * QSTR + kh * 16];
#pragma unroll
      for (int j = 0; j < 8; ++j) bfr[j] = *(const u32*)(p + 2 * j);
      s0 = __builtin_amdgcn_wmma_f32_16x16x32_bf16(false, as_bf16x16(qa0), false,
                                                   as_bf16x16(bfr), (short)0, s0, false, false);
#pragma unroll
      for (int j = 0; j < 8; ++j) bfr[j] = *(const u32*)(p + 32 + 2 * j);
      s0 = __builtin_amdgcn_wmma_f32_16x16x32_bf16(false, as_bf16x16(qa1), false,
                                                   as_bf16x16(bfr), (short)0, s0, false, false);
    }
    {
      const u16* p = &Ks[(16 + rl) * QSTR + kh * 16];
#pragma unroll
      for (int j = 0; j < 8; ++j) bfr[j] = *(const u32*)(p + 2 * j);
      s1 = __builtin_amdgcn_wmma_f32_16x16x32_bf16(false, as_bf16x16(qa0), false,
                                                   as_bf16x16(bfr), (short)0, s1, false, false);
#pragma unroll
      for (int j = 0; j < 8; ++j) bfr[j] = *(const u32*)(p + 32 + 2 * j);
      s1 = __builtin_amdgcn_wmma_f32_16x16x32_bf16(false, as_bf16x16(qa1), false,
                                                   as_bf16x16(bfr), (short)0, s1, false, false);
    }

    // online softmax; rows live in 16-lane groups -> xor-shuffle reductions
    float fac[8];
#pragma unroll
    for (int j = 0; j < 8; ++j) {
      float v0 = s0[j] * scale, v1 = s1[j] * scale;
      float t = fmaxf(v0, v1);
      t = fmaxf(t, __shfl_xor(t, 1, 32));
      t = fmaxf(t, __shfl_xor(t, 2, 32));
      t = fmaxf(t, __shfl_xor(t, 4, 32));
      t = fmaxf(t, __shfl_xor(t, 8, 32));
      float mn = fmaxf(mrow[j], t);
      float f = __expf(mrow[j] - mn);
      float p0 = __expf(v0 - mn), p1 = __expf(v1 - mn);
      float rs = p0 + p1;
      rs += __shfl_xor(rs, 1, 32);
      rs += __shfl_xor(rs, 2, 32);
      rs += __shfl_xor(rs, 4, 32);
      rs += __shfl_xor(rs, 8, 32);
      lrow[j] = lrow[j] * f + rs;
      mrow[j] = mn;
      fac[j] = f;
      const int row = j + 8 * kh;
      Ps[row * VSTR + rl] = f2bf(p0);
      Ps[row * VSTR + 16 + rl] = f2bf(p1);
    }
#pragma unroll
    for (int t = 0; t < 4; ++t)
#pragma unroll
      for (int j = 0; j < 8; ++j) O[t][j] = O[t][j] * fac[j];

    // P (16x32) as A-frag from LDS relayout
    u32 pf[8];
    {
      const u16* p = &Ps[rl * VSTR];
#pragma unroll
      for (int j = 0; j < 4; ++j) {
        pf[j]     = *(const u32*)(p + kh * 16 + 2 * j);
        pf[4 + j] = *(const u32*)(p + 8 + kh * 16 + 2 * j);
      }
    }
    // O[t] += P * V
#pragma unroll
    for (int t = 0; t < 4; ++t) {
      const u16* p = &Vt[(t * 16 + rl) * VSTR + kh * 16];
      u32 vf[8];
#pragma unroll
      for (int j = 0; j < 8; ++j) vf[j] = *(const u32*)(p + 2 * j);
      O[t] = __builtin_amdgcn_wmma_f32_16x16x32_bf16(false, as_bf16x16(pf), false,
                                                     as_bf16x16(vf), (short)0, O[t], false, false);
    }
  }

  // normalize and store ctx (bf16, [B,S,H,DK] == [B*S, D] layout)
#pragma unroll
  for (int t = 0; t < 4; ++t) {
#pragma unroll
    for (int j = 0; j < 8; ++j) {
      float v = O[t][j] / lrow[j];
      size_t off = base + (size_t)(qbase + j + 8 * kh) * DD + t * 16 + rl;
      Ctx[off] = f2bf(v);
    }
  }
}

// ---------- host orchestration ----------
extern "C" void kernel_launch(void* const* d_in, const int* in_sizes, int n_in,
                              void* d_out, int out_size, void* d_ws, size_t ws_size,
                              hipStream_t stream) {
  (void)in_sizes; (void)n_in; (void)out_size; (void)ws_size;
  const float* x  = (const float*)d_in[0];
  // d_in[1] src_mask: no effect in reference (non-in-place masked_fill) -> unused
  const float* Wq = (const float*)d_in[2];  const float* bq = (const float*)d_in[3];
  const float* Wk = (const float*)d_in[4];  const float* bk = (const float*)d_in[5];
  const float* Wv = (const float*)d_in[6];  const float* bv = (const float*)d_in[7];
  const float* Wo = (const float*)d_in[8];  const float* bo = (const float*)d_in[9];
  const float* W1 = (const float*)d_in[10]; const float* b1 = (const float*)d_in[11];
  const float* W2 = (const float*)d_in[12]; const float* b2 = (const float*)d_in[13];
  const float* a1 = (const float*)d_in[14]; const float* g1 = (const float*)d_in[15];
  const float* a2 = (const float*)d_in[16]; const float* g2 = (const float*)d_in[17];
  float* out = (float*)d_out;

  // workspace carve-up
  char* w = (char*)d_ws;
  auto alloc_u16 = [&](size_t n) { u16* p = (u16*)w; w += n * sizeof(u16); return p; };
  auto alloc_f32 = [&](size_t n) { float* p = (float*)w; w += n * sizeof(float); return p; };
  const size_t DxD = (size_t)DD * DD;           // 1M
  const size_t DxF = (size_t)DD * DFFV;         // 4M
  const size_t Act = (size_t)MM * DD;           // 4M
  const size_t ActF = (size_t)MM * DFFV;        // 16M
  u16* wq_b = alloc_u16(DxD);
  u16* wk_b = alloc_u16(DxD);
  u16* wv_b = alloc_u16(DxD);
  u16* wo_b = alloc_u16(DxD);
  u16* w1_b = alloc_u16(DxF);
  u16* w2_b = alloc_u16(DxF);
  u16* h1   = alloc_u16(Act);
  u16* qb   = alloc_u16(Act);
  u16* kb   = alloc_u16(Act);
  u16* vb   = alloc_u16(Act);
  u16* ctxb = alloc_u16(Act);
  u16* h2   = alloc_u16(Act);
  u16* ffb  = alloc_u16(ActF);
  float* x1 = alloc_f32(Act);

  // weight conversions (8 elems/thread)
  cvt_bf16_kernel<<<(int)((DxD / 8 + 255) / 256), 256, 0, stream>>>(Wq, wq_b, (int)(DxD / 8));
  cvt_bf16_kernel<<<(int)((DxD / 8 + 255) / 256), 256, 0, stream>>>(Wk, wk_b, (int)(DxD / 8));
  cvt_bf16_kernel<<<(int)((DxD / 8 + 255) / 256), 256, 0, stream>>>(Wv, wv_b, (int)(DxD / 8));
  cvt_bf16_kernel<<<(int)((DxD / 8 + 255) / 256), 256, 0, stream>>>(Wo, wo_b, (int)(DxD / 8));
  cvt_bf16_kernel<<<(int)((DxF / 8 + 255) / 256), 256, 0, stream>>>(W1, w1_b, (int)(DxF / 8));
  cvt_bf16_kernel<<<(int)((DxF / 8 + 255) / 256), 256, 0, stream>>>(W2, w2_b, (int)(DxF / 8));

  // LN1
  ln_bf16_kernel<<<MM, 256, 0, stream>>>(x, a1, g1, h1, DD);

  // QKV projections (bf16 WMMA, f32 accumulate, bf16 out)
  dim3 gD(DD / 128, MM / 128);
  gemm_bf16_kernel<false, false, false, true><<<gD, 256, 0, stream>>>(
      h1, wq_b, bq, nullptr, nullptr, qb, MM, DD, DD);
  gemm_bf16_kernel<false, false, false, true><<<gD, 256, 0, stream>>>(
      h1, wk_b, bk, nullptr, nullptr, kb, MM, DD, DD);
  gemm_bf16_kernel<false, false, false, true><<<gD, 256, 0, stream>>>(
      h1, wv_b, bv, nullptr, nullptr, vb, MM, DD, DD);

  // flash attention (no score materialization)
  flash_attn_kernel<<<BB * HH * (SS / 16), 32, 0, stream>>>(qb, kb, vb, ctxb);

  // output projection + residual (fp32 x1)
  gemm_bf16_kernel<false, true, true, false><<<gD, 256, 0, stream>>>(
      ctxb, wo_b, bo, x, x1, nullptr, MM, DD, DD);

  // LN2
  ln_bf16_kernel<<<MM, 256, 0, stream>>>(x1, a2, g2, h2, DD);

  // FFN
  dim3 gF(DFFV / 128, MM / 128);
  gemm_bf16_kernel<true, false, false, true><<<gF, 256, 0, stream>>>(
      h2, w1_b, b1, nullptr, nullptr, ffb, MM, DFFV, DD);
  gemm_bf16_kernel<false, true, true, false><<<gD, 256, 0, stream>>>(
      ffb, w2_b, b2, x1, out, nullptr, MM, DD, DFFV);
}
